// LSTMAutoEncoder_13657996001638
// MI455X (gfx1250) — matrix operand order repaired and checked by reading, not compiled, not run on hardware
//
#include <hip/hip_runtime.h>

typedef __attribute__((ext_vector_type(16))) __bf16 v16bf;
typedef __attribute__((ext_vector_type(8)))  __bf16 v8bf;
typedef __attribute__((ext_vector_type(8)))  float  v8f;

#define B_   128
#define T_   512
#define I_   75
#define H_   512
#define L_   64

// LDS strides (elements), padded to spread banks
#define HSTR  520   // 512 + 8 bf16
#define XSTRF 100   // 96 + 4 f32 (x staged as raw f32 by async copy)
#define DSTR  104   // 96 + 8 bf16 (decoder hidden padded 75->96)

static __device__ __forceinline__ float fast_tanh(float x){
#if __has_builtin(__builtin_amdgcn_tanhf)
  return __builtin_amdgcn_tanhf(x);
#else
  return tanhf(x);
#endif
}
// sigmoid via hw tanh: 1/(1+e^-x) = 0.5*tanh(x/2) + 0.5
static __device__ __forceinline__ float sigm(float v){ return 0.5f*fast_tanh(0.5f*v) + 0.5f; }

// ---- CDNA5 async global->LDS copy (ASYNCcnt path) ----
static __device__ __forceinline__ void async_ld_f32(unsigned lds_off, const float* gaddr){
  // VDST = VGPR holding LDS byte offset, VADDR = 64-bit global address
  asm volatile("global_load_async_to_lds_b32 %0, %1, off" :: "v"(lds_off), "v"(gaddr) : "memory");
}
static __device__ __forceinline__ void wait_async(){
  asm volatile("s_wait_asynccnt 0x0" ::: "memory");
}
static __device__ __forceinline__ unsigned lds_off_of(const void* p){
  // addrspacecast(LDS->flat) keeps the LDS offset in the low 32 bits
  return (unsigned)(size_t)p;
}

// A-fragment (16x32 bf16) from bf16 LDS row-major [16][stride].
// lanes 0-15 -> rows 0-15, K = k0+{0..7,16..23}; lanes 16-31 -> K = k0+{8..15,24..31}.
static __device__ __forceinline__ v16bf lds_a_frag(const __bf16* base, int stride, int k0, int lane){
  int row  = lane & 15;
  int koff = (lane < 16) ? 0 : 8;
  const __bf16* p = base + row*stride + k0 + koff;
  v8bf lo = *(const v8bf*)(p);
  v8bf hi = *(const v8bf*)(p + 16);
  return __builtin_shufflevector(lo, hi, 0,1,2,3,4,5,6,7,8,9,10,11,12,13,14,15);
}

// A-fragment from f32 LDS (converted to bf16 on the fly) -- used for async-staged x
static __device__ __forceinline__ v16bf lds_a_frag_f32(const float* base, int stride, int k0, int lane){
  int row  = lane & 15;
  int koff = (lane < 16) ? 0 : 8;
  const float4* p0 = (const float4*)(base + row*stride + k0 + koff);
  const float4* p1 = (const float4*)(base + row*stride + k0 + koff + 16);
  float4 x0 = p0[0], x1 = p0[1], y0 = p1[0], y1 = p1[1];
  v16bf a;
  a[0]=(__bf16)x0.x;  a[1]=(__bf16)x0.y;  a[2]=(__bf16)x0.z;  a[3]=(__bf16)x0.w;
  a[4]=(__bf16)x1.x;  a[5]=(__bf16)x1.y;  a[6]=(__bf16)x1.z;  a[7]=(__bf16)x1.w;
  a[8]=(__bf16)y0.x;  a[9]=(__bf16)y0.y;  a[10]=(__bf16)y0.z; a[11]=(__bf16)y0.w;
  a[12]=(__bf16)y1.x; a[13]=(__bf16)y1.y; a[14]=(__bf16)y1.z; a[15]=(__bf16)y1.w;
  return a;
}

// B-fragment (32x16 bf16) from global packed weights W[nrow][KP] row-major.
static __device__ __forceinline__ v16bf glb_b_frag(const __bf16* W, int KP, int nrow, int k0, int lane){
  int koff = (lane < 16) ? 0 : 8;
  const __bf16* p = W + (size_t)nrow*KP + k0 + koff;
  v8bf lo = *(const v8bf*)(p);
  v8bf hi = *(const v8bf*)(p + 16);
  return __builtin_shufflevector(lo, hi, 0,1,2,3,4,5,6,7,8,9,10,11,12,13,14,15);
}

static __device__ __forceinline__ v8f wmma_bf16(v16bf a, v16bf b, v8f c){
  return __builtin_amdgcn_wmma_f32_16x16x32_bf16(false, a, false, b, (short)0, c, false, false);
}

// ---------------------------------------------------------------- utilities

__global__ void k_zero_bf16(__bf16* p, int n){
  int i = blockIdx.x*blockDim.x + threadIdx.x;
  if (i < n) p[i] = (__bf16)0.0f;
}

__global__ void k_pack_bf16(__bf16* dst, int dstStride, int colOff, const float* src, int N, int K){
  int i = blockIdx.x*blockDim.x + threadIdx.x;
  if (i < N*K){
    int r = i / K, k = i - r*K;
    dst[(size_t)r*dstStride + colOff + k] = (__bf16)src[i];
  }
}

__global__ void k_bias_pad(float* dst, const float* a, const float* b, int N, int NP){
  int i = blockIdx.x*blockDim.x + threadIdx.x;
  if (i < NP) dst[i] = (i < N) ? (a[i] + b[i]) : 0.0f;
}

// ------------------------------------------------- generic WMMA GEMM (small)
// C[M x Nout] f32 = A[M x K] f32 (cvt->bf16) * W[NP x KP]^T bf16 + bias[NP]
// REQUIRES K == KP (multiple of 32). grid = (M/16, NP/64), block = 32.
__global__ __launch_bounds__(32) void wmma_gemm(const float* __restrict__ A, int K,
                                                const __bf16* __restrict__ W,
                                                const float* __restrict__ bias,
                                                float* __restrict__ C, int ldc, int Nout){
  int lane = threadIdx.x & 31;
  int ln   = lane & 15;
  int mb   = blockIdx.x * 16;
  int nb   = blockIdx.y * 64;

  v8f acc[4];
  #pragma unroll
  for (int tix = 0; tix < 4; ++tix){
    float bv = bias[nb + tix*16 + ln];
    #pragma unroll
    for (int e = 0; e < 8; ++e) acc[tix][e] = bv;
  }

  const float* Arow = A + (size_t)(mb + ln)*K;
  int koff = (lane < 16) ? 0 : 8;
  for (int k0 = 0; k0 < K; k0 += 32){
    const float4* p0 = (const float4*)(Arow + k0 + koff);
    const float4* p1 = (const float4*)(Arow + k0 + koff + 16);
    float4 x0 = p0[0], x1 = p0[1], y0 = p1[0], y1 = p1[1];
    v16bf a;
    a[0]=(__bf16)x0.x;  a[1]=(__bf16)x0.y;  a[2]=(__bf16)x0.z;  a[3]=(__bf16)x0.w;
    a[4]=(__bf16)x1.x;  a[5]=(__bf16)x1.y;  a[6]=(__bf16)x1.z;  a[7]=(__bf16)x1.w;
    a[8]=(__bf16)y0.x;  a[9]=(__bf16)y0.y;  a[10]=(__bf16)y0.z; a[11]=(__bf16)y0.w;
    a[12]=(__bf16)y1.x; a[13]=(__bf16)y1.y; a[14]=(__bf16)y1.z; a[15]=(__bf16)y1.w;
    #pragma unroll
    for (int tix = 0; tix < 4; ++tix){
      v16bf b = glb_b_frag(W, K, nb + tix*16 + ln, k0, lane);
      acc[tix] = wmma_bf16(a, b, acc[tix]);
    }
  }

  #pragma unroll
  for (int tix = 0; tix < 4; ++tix){
    int col = nb + tix*16 + ln;
    if (col < Nout){
      #pragma unroll
      for (int e = 0; e < 8; ++e){
        int r = mb + ((lane < 16) ? e : e + 8);
        C[(size_t)r*ldc + col] = acc[tix][e];
      }
    }
  }
}

// ------------------------------------------------- fused 2-layer encoder scan
// grid = 8 (B/16), block = 512 (16 waves). Wave w owns hidden units [32w,32w+32).
// W0 = [Whh0 | Wih0(pad 96)] -> [2048][608]; W1 = [Whh1 | Wih1] -> [2048][1024]
__global__ __launch_bounds__(512) void enc_scan(const float* __restrict__ x,
                                                const __bf16* __restrict__ W0,
                                                const float*  __restrict__ b0,
                                                const __bf16* __restrict__ W1,
                                                const float*  __restrict__ b1,
                                                float* __restrict__ hlast){
  __shared__ __attribute__((aligned(32))) __bf16 h0[16*HSTR];
  __shared__ __attribute__((aligned(32))) __bf16 h1[16*HSTR];
  __shared__ __attribute__((aligned(32))) float  xa[16*XSTRF];

  const int tid   = threadIdx.x;
  const int lane  = tid & 31;
  const int wave  = tid >> 5;
  const int ln    = lane & 15;
  const int jb    = wave * 32;
  const int bbase = blockIdx.x * 16;

  for (int i = tid; i < 16*HSTR; i += 512){ h0[i] = (__bf16)0.0f; h1[i] = (__bf16)0.0f; }
  for (int i = tid; i < 16*XSTRF; i += 512){ xa[i] = 0.0f; }   // pad cols stay 0 forever
  __syncthreads();

  // async-stage x_0 into LDS (raw f32)
  for (int i = tid; i < 16*I_; i += 512){
    int r = i / I_, k = i - r*I_;
    async_ld_f32(lds_off_of(&xa[r*XSTRF + k]), x + ((size_t)(bbase + r)*T_ + 0)*I_ + k);
  }

  float bia0[8], bia1[8];
  #pragma unroll
  for (int g = 0; g < 4; ++g)
    #pragma unroll
    for (int hh = 0; hh < 2; ++hh){
      bia0[g*2+hh] = b0[g*H_ + jb + hh*16 + ln];
      bia1[g*2+hh] = b1[g*H_ + jb + hh*16 + ln];
    }

  v8f c0[2], c1[2];
  #pragma unroll
  for (int hh = 0; hh < 2; ++hh)
    #pragma unroll
    for (int e = 0; e < 8; ++e){ c0[hh][e] = 0.0f; c1[hh][e] = 0.0f; }

  wait_async();
  __syncthreads();

  for (int t = 0; t < T_; ++t){
    // ---------------- layer 0:  g = b0 + [h0 | x_t] * [Whh0 | Wih0]^T
    v8f acc[8];
    #pragma unroll
    for (int q = 0; q < 8; ++q)
      #pragma unroll
      for (int e = 0; e < 8; ++e) acc[q][e] = bia0[q];

    for (int kc = 0; kc < 16; ++kc){              // recurrent K (h0, 512)
      v16bf a = lds_a_frag(h0, HSTR, kc*32, lane);
      #pragma unroll
      for (int g = 0; g < 4; ++g)
        #pragma unroll
        for (int hh = 0; hh < 2; ++hh){
          v16bf b = glb_b_frag(W0, 608, g*H_ + jb + hh*16 + ln, kc*32, lane);
          acc[g*2+hh] = wmma_bf16(a, b, acc[g*2+hh]);
        }
    }
    for (int kc = 0; kc < 3; ++kc){               // input K (x_t, 96 padded)
      v16bf a = lds_a_frag_f32(xa, XSTRF, kc*32, lane);
      #pragma unroll
      for (int g = 0; g < 4; ++g)
        #pragma unroll
        for (int hh = 0; hh < 2; ++hh){
          v16bf b = glb_b_frag(W0, 608, g*H_ + jb + hh*16 + ln, 512 + kc*32, lane);
          acc[g*2+hh] = wmma_bf16(a, b, acc[g*2+hh]);
        }
    }

    v8f h0n[2];
    #pragma unroll
    for (int hh = 0; hh < 2; ++hh)
      #pragma unroll
      for (int e = 0; e < 8; ++e){
        float ig = sigm(acc[0+hh][e]);
        float fg = sigm(acc[2+hh][e]);
        float gg = fast_tanh(acc[4+hh][e]);
        float og = sigm(acc[6+hh][e]);
        float cn = fg*c0[hh][e] + ig*gg;
        c0[hh][e] = cn;
        h0n[hh][e] = og * fast_tanh(cn);
      }

    __syncthreads();   // all reads of h0/xa done
    #pragma unroll
    for (int hh = 0; hh < 2; ++hh)
      #pragma unroll
      for (int e = 0; e < 8; ++e){
        int r = (lane < 16) ? e : e + 8;
        h0[r*HSTR + jb + hh*16 + ln] = (__bf16)h0n[hh][e];
      }
    if (t + 1 < T_){
      // async-stage x_{t+1}; overlaps with the layer-1 WMMA pass below
      for (int i = tid; i < 16*I_; i += 512){
        int r = i / I_, k = i - r*I_;
        async_ld_f32(lds_off_of(&xa[r*XSTRF + k]), x + ((size_t)(bbase + r)*T_ + (t+1))*I_ + k);
      }
    }
    __syncthreads();   // new h0 visible

    // ---------------- layer 1:  g = b1 + [h1 | h0_new] * [Whh1 | Wih1]^T
    #pragma unroll
    for (int q = 0; q < 8; ++q)
      #pragma unroll
      for (int e = 0; e < 8; ++e) acc[q][e] = bia1[q];

    for (int kc = 0; kc < 16; ++kc){
      v16bf a = lds_a_frag(h1, HSTR, kc*32, lane);
      #pragma unroll
      for (int g = 0; g < 4; ++g)
        #pragma unroll
        for (int hh = 0; hh < 2; ++hh){
          v16bf b = glb_b_frag(W1, 1024, g*H_ + jb + hh*16 + ln, kc*32, lane);
          acc[g*2+hh] = wmma_bf16(a, b, acc[g*2+hh]);
        }
    }
    for (int kc = 0; kc < 16; ++kc){
      v16bf a = lds_a_frag(h0, HSTR, kc*32, lane);
      #pragma unroll
      for (int g = 0; g < 4; ++g)
        #pragma unroll
        for (int hh = 0; hh < 2; ++hh){
          v16bf b = glb_b_frag(W1, 1024, g*H_ + jb + hh*16 + ln, 512 + kc*32, lane);
          acc[g*2+hh] = wmma_bf16(a, b, acc[g*2+hh]);
        }
    }

    v8f h1n[2];
    #pragma unroll
    for (int hh = 0; hh < 2; ++hh)
      #pragma unroll
      for (int e = 0; e < 8; ++e){
        float ig = sigm(acc[0+hh][e]);
        float fg = sigm(acc[2+hh][e]);
        float gg = fast_tanh(acc[4+hh][e]);
        float og = sigm(acc[6+hh][e]);
        float cn = fg*c1[hh][e] + ig*gg;
        c1[hh][e] = cn;
        h1n[hh][e] = og * fast_tanh(cn);
      }

    __syncthreads();   // all reads of h1 done
    #pragma unroll
    for (int hh = 0; hh < 2; ++hh)
      #pragma unroll
      for (int e = 0; e < 8; ++e){
        int r = (lane < 16) ? e : e + 8;
        h1[r*HSTR + jb + hh*16 + ln] = (__bf16)h1n[hh][e];
      }
    wait_async();      // x_{t+1} staged before next iteration's reads
    __syncthreads();
  }

  for (int i = tid; i < 16*H_; i += 512){
    int r = i >> 9, c = i & 511;
    hlast[(size_t)(bbase + r)*H_ + c] = (float)h1[r*HSTR + c];
  }
}

// ------------------------------------------------- fused 2-layer decoder scan
// hidden=75 (padded 96). grid = 8, block = 160 (5 waves). Wave w owns units [16w,16w+16).
__global__ __launch_bounds__(160) void dec_scan(const float* __restrict__ dg0,
                                                const __bf16* __restrict__ Wd0,
                                                const __bf16* __restrict__ Wd1,
                                                const float*  __restrict__ b1,
                                                float* __restrict__ out){
  __shared__ __attribute__((aligned(32))) __bf16 h0[16*DSTR];
  __shared__ __attribute__((aligned(32))) __bf16 h1[16*DSTR];

  const int tid   = threadIdx.x;
  const int lane  = tid & 31;
  const int wave  = tid >> 5;
  const int ln    = lane & 15;
  const int jb    = wave * 16;
  const int j     = jb + ln;                 // hidden unit (may be pad >= 75)
  const int bbase = blockIdx.x * 16;

  for (int i = tid; i < 16*DSTR; i += 160){ h0[i] = (__bf16)0.0f; h1[i] = (__bf16)0.0f; }

  v8f d0init[4];
  float bv1[4];
  #pragma unroll
  for (int g = 0; g < 4; ++g){
    #pragma unroll
    for (int e = 0; e < 8; ++e){
      int r = (lane < 16) ? e : e + 8;
      d0init[g][e] = dg0[(size_t)(bbase + r)*320 + g*I_ + j];
    }
    bv1[g] = b1[g*I_ + j];
  }

  v8f c0, c1;
  #pragma unroll
  for (int e = 0; e < 8; ++e){ c0[e] = 0.0f; c1[e] = 0.0f; }

  __syncthreads();

  for (int t = 0; t < T_; ++t){
    // layer 0: g = dg0 + h0 * Whh0^T
    v8f acc[4];
    #pragma unroll
    for (int g = 0; g < 4; ++g) acc[g] = d0init[g];
    for (int kc = 0; kc < 3; ++kc){
      v16bf a = lds_a_frag(h0, DSTR, kc*32, lane);
      #pragma unroll
      for (int g = 0; g < 4; ++g){
        v16bf b = glb_b_frag(Wd0, 96, g*I_ + j, kc*32, lane);
        acc[g] = wmma_bf16(a, b, acc[g]);
      }
    }
    v8f h0n;
    #pragma unroll
    for (int e = 0; e < 8; ++e){
      float ig = sigm(acc[0][e]);
      float fg = sigm(acc[1][e]);
      float gg = fast_tanh(acc[2][e]);
      float og = sigm(acc[3][e]);
      float cn = fg*c0[e] + ig*gg;
      c0[e] = cn;
      h0n[e] = og * fast_tanh(cn);
    }
    __syncthreads();
    #pragma unroll
    for (int e = 0; e < 8; ++e){
      int r = (lane < 16) ? e : e + 8;
      h0[r*DSTR + j] = (j < I_) ? (__bf16)h0n[e] : (__bf16)0.0f;
    }
    __syncthreads();

    // layer 1: g = b1 + [h1 | h0_new] * [Whh1 | Wih1]^T
    #pragma unroll
    for (int g = 0; g < 4; ++g)
      #pragma unroll
      for (int e = 0; e < 8; ++e) acc[g][e] = bv1[g];
    for (int kc = 0; kc < 3; ++kc){
      v16bf a = lds_a_frag(h1, DSTR, kc*32, lane);
      #pragma unroll
      for (int g = 0; g < 4; ++g){
        v16bf b = glb_b_frag(Wd1, 192, g*I_ + j, kc*32, lane);
        acc[g] = wmma_bf16(a, b, acc[g]);
      }
    }
    for (int kc = 0; kc < 3; ++kc){
      v16bf a = lds_a_frag(h0, DSTR, kc*32, lane);
      #pragma unroll
      for (int g = 0; g < 4; ++g){
        v16bf b = glb_b_frag(Wd1, 192, g*I_ + j, 96 + kc*32, lane);
        acc[g] = wmma_bf16(a, b, acc[g]);
      }
    }
    v8f h1n;
    #pragma unroll
    for (int e = 0; e < 8; ++e){
      float ig = sigm(acc[0][e]);
      float fg = sigm(acc[1][e]);
      float gg = fast_tanh(acc[2][e]);
      float og = sigm(acc[3][e]);
      float cn = fg*c1[e] + ig*gg;
      c1[e] = cn;
      h1n[e] = og * fast_tanh(cn);
    }
    __syncthreads();
    #pragma unroll
    for (int e = 0; e < 8; ++e){
      int r = (lane < 16) ? e : e + 8;
      h1[r*DSTR + j] = (j < I_) ? (__bf16)h1n[e] : (__bf16)0.0f;
      if (j < I_) out[((size_t)(bbase + r)*T_ + t)*I_ + j] = h1n[e];
    }
    __syncthreads();
  }
}

// ---------------------------------------------------------------- launch

extern "C" void kernel_launch(void* const* d_in, const int* in_sizes, int n_in,
                              void* d_out, int out_size, void* d_ws, size_t ws_size,
                              hipStream_t stream){
  const float* x     = (const float*)d_in[0];
  const float* eWih0 = (const float*)d_in[1];
  const float* eWhh0 = (const float*)d_in[2];
  const float* ebih0 = (const float*)d_in[3];
  const float* ebhh0 = (const float*)d_in[4];
  const float* eWih1 = (const float*)d_in[5];
  const float* eWhh1 = (const float*)d_in[6];
  const float* ebih1 = (const float*)d_in[7];
  const float* ebhh1 = (const float*)d_in[8];
  const float* efcW  = (const float*)d_in[9];
  const float* efcb  = (const float*)d_in[10];
  const float* dfcW  = (const float*)d_in[11];
  const float* dfcb  = (const float*)d_in[12];
  const float* dWih0 = (const float*)d_in[13];
  const float* dWhh0 = (const float*)d_in[14];
  const float* dbih0 = (const float*)d_in[15];
  const float* dbhh0 = (const float*)d_in[16];
  const float* dWih1 = (const float*)d_in[17];
  const float* dWhh1 = (const float*)d_in[18];
  const float* dbih1 = (const float*)d_in[19];
  const float* dbhh1 = (const float*)d_in[20];

  char* ws = (char*)d_ws;
  size_t off = 0;
  auto take = [&](size_t bytes) -> char* {
    char* p = ws + off;
    off += (bytes + 255) & ~(size_t)255;
    return p;
  };

  __bf16* Wc0  = (__bf16*)take((size_t)2048*608*2);   // [Whh0 | Wih0 pad96]
  __bf16* Wc1  = (__bf16*)take((size_t)2048*1024*2);  // [Whh1 | Wih1]
  __bf16* Wefc = (__bf16*)take((size_t)64*512*2);
  __bf16* Wdfc = (__bf16*)take((size_t)512*64*2);
  __bf16* Wdi0 = (__bf16*)take((size_t)320*512*2);    // d_Wih0 padded rows
  __bf16* Wd0  = (__bf16*)take((size_t)320*96*2);     // d_Whh0 padded
  __bf16* Wd1  = (__bf16*)take((size_t)320*192*2);    // [d_Whh1 | d_Wih1] padded
  float*  be0  = (float*)take((size_t)2048*4);
  float*  be1  = (float*)take((size_t)2048*4);
  float*  bd0  = (float*)take((size_t)320*4);
  float*  bd1  = (float*)take((size_t)320*4);
  float*  hL   = (float*)take((size_t)128*512*4);
  float*  z    = (float*)take((size_t)128*64*4);
  float*  emb  = (float*)take((size_t)128*512*4);
  float*  dg0  = (float*)take((size_t)128*320*4);

  auto g1 = [](int n){ return (n + 255) / 256; };

  // zero padded weight buffers
  k_zero_bf16<<<g1(2048*608), 256, 0, stream>>>(Wc0, 2048*608);
  k_zero_bf16<<<g1(320*512),  256, 0, stream>>>(Wdi0, 320*512);
  k_zero_bf16<<<g1(320*96),   256, 0, stream>>>(Wd0, 320*96);
  k_zero_bf16<<<g1(320*192),  256, 0, stream>>>(Wd1, 320*192);

  // pack / convert weights to bf16
  k_pack_bf16<<<g1(2048*512), 256, 0, stream>>>(Wc0, 608, 0,   eWhh0, 2048, 512);
  k_pack_bf16<<<g1(2048*75),  256, 0, stream>>>(Wc0, 608, 512, eWih0, 2048, 75);
  k_pack_bf16<<<g1(2048*512), 256, 0, stream>>>(Wc1, 1024, 0,   eWhh1, 2048, 512);
  k_pack_bf16<<<g1(2048*512), 256, 0, stream>>>(Wc1, 1024, 512, eWih1, 2048, 512);
  k_pack_bf16<<<g1(64*512),   256, 0, stream>>>(Wefc, 512, 0, efcW, 64, 512);
  k_pack_bf16<<<g1(512*64),   256, 0, stream>>>(Wdfc, 64, 0,  dfcW, 512, 64);
  k_pack_bf16<<<g1(300*512),  256, 0, stream>>>(Wdi0, 512, 0, dWih0, 300, 512);
  k_pack_bf16<<<g1(300*75),   256, 0, stream>>>(Wd0, 96, 0,   dWhh0, 300, 75);
  k_pack_bf16<<<g1(300*75),   256, 0, stream>>>(Wd1, 192, 0,  dWhh1, 300, 75);
  k_pack_bf16<<<g1(300*75),   256, 0, stream>>>(Wd1, 192, 96, dWih1, 300, 75);

  // combined biases
  k_bias_pad<<<g1(2048), 256, 0, stream>>>(be0, ebih0, ebhh0, 2048, 2048);
  k_bias_pad<<<g1(2048), 256, 0, stream>>>(be1, ebih1, ebhh1, 2048, 2048);
  k_bias_pad<<<g1(320),  256, 0, stream>>>(bd0, dbih0, dbhh0, 300, 320);
  k_bias_pad<<<g1(320),  256, 0, stream>>>(bd1, dbih1, dbhh1, 300, 320);

  // encoder: fused 2-layer scan over T, produces final top hidden state
  enc_scan<<<8, 512, 0, stream>>>(x, Wc0, be0, Wc1, be1, hL);

  // bottleneck: z = hL @ efc_W^T + efc_b ; emb = z @ dfc_W^T + dfc_b
  wmma_gemm<<<dim3(8, 1), 32, 0, stream>>>(hL, 512, Wefc, efcb, z, 64, 64);
  wmma_gemm<<<dim3(8, 8), 32, 0, stream>>>(z, 64, Wdfc, dfcb, emb, 512, 512);

  // decoder layer-0 input contribution (constant over T): dg0 = emb @ d_Wih0^T + b
  wmma_gemm<<<dim3(8, 5), 32, 0, stream>>>(emb, 512, Wdi0, bd0, dg0, 320, 320);

  // decoder: fused 2-layer scan, writes output [B,T,75] f32
  dec_scan<<<8, 160, 0, stream>>>(dg0, Wd0, Wd1, bd1, (float*)d_out);
}
